// Block_31147102831158
// MI455X (gfx1250) — compile-verified, matrix operand order checked by reading, not run on hardware
//
#include <hip/hip_runtime.h>
#include <hip/hip_bf16.h>
#include <math.h>

#define B_DIM 8
#define L_DIM 8192
#define D_DIM 256
#define H_DIM 1024
#define NC    4096   /* complex FFT length = L/2 */

typedef __attribute__((ext_vector_type(16))) __bf16 v16bf;
typedef __attribute__((ext_vector_type(8)))  float  v8f;
typedef unsigned short u16;

struct U4x2 { uint4 a, b; };
union AccU { v8f v; float f[8]; };

// ---- CDNA5 async global->LDS staging (signature: (v4i AS1*, v4i AS3*, imm, imm))
#if __has_builtin(__builtin_amdgcn_global_load_async_to_lds_b128)
#define HAVE_ASYNC_LDS 1
typedef __attribute__((ext_vector_type(4))) int v4i;
typedef __attribute__((address_space(1))) v4i* as1_v4i;
typedef __attribute__((address_space(3))) v4i* as3_v4i;
#define ASYNC_CP16(gsrc, ldst)                                              \
  __builtin_amdgcn_global_load_async_to_lds_b128(                           \
      (as1_v4i)(gsrc), (as3_v4i)(ldst), 0, 0)
__device__ __forceinline__ void async_wait0() {
#if __has_builtin(__builtin_amdgcn_s_wait_asynccnt)
  __builtin_amdgcn_s_wait_asynccnt(0);
#else
  asm volatile("s_wait_asynccnt 0x0" ::: "memory");
#endif
}
#endif

__device__ __forceinline__ u16 f2bf(float f) {
  unsigned u = __float_as_uint(f);
  u += 0x7FFFu + ((u >> 16) & 1u);
  return (u16)(u >> 16);
}

__device__ __forceinline__ float wave_sum(float v) {
  for (int o = 16; o > 0; o >>= 1) v += __shfl_xor(v, o, 32);
  return v;
}

// ---------------------------------------------------------------------------
// K0: weights -> bf16, transposed so each output column is K-contiguous
// ---------------------------------------------------------------------------
__global__ __launch_bounds__(256) void prep_w_kernel(
    const float* __restrict__ w1, const float* __restrict__ w2,
    u16* __restrict__ W1t, u16* __restrict__ W2t) {
  int i = blockIdx.x * 256 + threadIdx.x;          // 262144 elements each
  if (i < D_DIM * H_DIM) {
    int k1 = i >> 10, n1 = i & 1023;               // w1 (256 x 1024)
    W1t[(size_t)n1 * D_DIM + k1] = f2bf(w1[i]);
    int k2 = i >> 8, n2 = i & 255;                 // w2 (1024 x 256)
    W2t[(size_t)n2 * H_DIM + k2] = f2bf(w2[i]);
  }
}

// ---------------------------------------------------------------------------
// K1: LayerNorm1 + ReLU + eps, tile-transposed write (B,L,D) -> (B,D,L)
// ---------------------------------------------------------------------------
__global__ __launch_bounds__(256) void ln1_relu_T_kernel(
    const float* __restrict__ x, const float* __restrict__ g1,
    const float* __restrict__ b1, float* __restrict__ ht) {
  __shared__ float tile[32][D_DIM + 1];
  const int b = blockIdx.y, l0 = blockIdx.x * 32, t = threadIdx.x;
  const float* xb = x + ((size_t)b * L_DIM + l0) * D_DIM;
#pragma unroll 8
  for (int r = 0; r < 32; ++r) tile[r][t] = xb[r * D_DIM + t];
  __syncthreads();
  const int wid = t >> 5, lane = t & 31;
  for (int q = 0; q < 4; ++q) {
    const int r = wid * 4 + q;
    float s = 0.f, s2 = 0.f;
#pragma unroll
    for (int i = 0; i < 8; ++i) {
      float v = tile[r][lane + 32 * i];
      s += v; s2 += v * v;
    }
    s = wave_sum(s); s2 = wave_sum(s2);
    float mean = s * (1.f / D_DIM);
    float var  = s2 * (1.f / D_DIM) - mean * mean;
    float rstd = rsqrtf(var + 1e-5f);
#pragma unroll
    for (int i = 0; i < 8; ++i) {
      int d = lane + 32 * i;
      float v = (tile[r][d] - mean) * rstd * g1[d] + b1[d];
      tile[r][d] = fmaxf(v, 0.f) + 1e-6f;
    }
  }
  __syncthreads();
  float* dst = ht + ((size_t)b * D_DIM + t) * L_DIM + l0;   // thread t == d
#pragma unroll
  for (int j = 0; j < 32; j += 4) {
    float4 v = make_float4(tile[j][t], tile[j + 1][t], tile[j + 2][t], tile[j + 3][t]);
    *(float4*)(dst + j) = v;
  }
}

// ---------------------------------------------------------------------------
// K2: per-(b,d) rfft(8192) via packed 4096-pt complex FFT in LDS, spectral
// window-max filter, irfft via conjugate trick. In-place on ht. 2048 blocks.
// ---------------------------------------------------------------------------
__device__ __forceinline__ void fft_stages(float2* z, int t) {
  for (int s = 1; s <= 12; ++s) {
    const int m = 1 << s, half = m >> 1;
    for (int j = t; j < NC / 2; j += 256) {
      int pos = j & (half - 1);
      int i0 = ((j >> (s - 1)) << s) + pos;
      int i1 = i0 + half;
      float ss, cc;
      __sincosf(-6.283185307179586f * pos / m, &ss, &cc);
      float2 a = z[i0], bb = z[i1];
      float2 tt = { bb.x * cc - bb.y * ss, bb.x * ss + bb.y * cc };
      z[i0] = { a.x + tt.x, a.y + tt.y };
      z[i1] = { a.x - tt.x, a.y - tt.y };
    }
    __syncthreads();
  }
}

// rfft bin k (0..4096) from packed complex spectrum Z
__device__ __forceinline__ float2 spec_bin(const float2* z, int k) {
  float2 Zk = z[k & (NC - 1)];
  float2 Zn = z[(NC - k) & (NC - 1)];
  float2 Xe = { 0.5f * (Zk.x + Zn.x), 0.5f * (Zk.y - Zn.y) };
  float2 Dd = { Zk.x - Zn.x, Zk.y + Zn.y };
  float2 Xo = { 0.5f * Dd.y, -0.5f * Dd.x };
  float ss, cc;
  __sincosf(-3.14159265358979f * k * (1.f / NC), &ss, &cc);
  return { Xe.x + cc * Xo.x - ss * Xo.y, Xe.y + cc * Xo.y + ss * Xo.x };
}

__global__ __launch_bounds__(256) void fft_filter_kernel(float* __restrict__ ht) {
  __shared__ float2 z[NC];                 // 32 KB
  __shared__ float  mag[NC + 1];           // 16.4 KB
  __shared__ unsigned char keep[NC + 1];   // 4.1 KB
  const int t = threadIdx.x;
  float* sig = ht + (size_t)blockIdx.x * L_DIM;

  for (int n = t; n < NC; n += 256) {
    float2 v = ((const float2*)sig)[n];
    z[__brev((unsigned)n) >> 20] = v;
  }
  __syncthreads();
  fft_stages(z, t);

  for (int k = t; k <= NC; k += 256) {
    float2 X = spec_bin(z, k);
    mag[k] = sqrtf(X.x * X.x + X.y * X.y);
  }
  __syncthreads();

  // env = maxpool window 8, SAME (pad 3 lo / 4 hi) dominates scales 1/2/4
  for (int k = t; k <= NC; k += 256) {
    int lo = k - 3 < 0 ? 0 : k - 3;
    int hi = k + 4 > NC ? NC : k + 4;
    float mx = -1e30f;
    for (int j = lo; j <= hi; ++j) mx = fmaxf(mx, mag[j]);
    keep[k] = (mag[k] >= mx) || (k < 3);
  }
  __syncthreads();

  for (int k = t; k <= NC / 2; k += 256) {
    int nk = NC - k;
    float2 Xk = spec_bin(z, k);  if (!keep[k])  { Xk.x = 0.f; Xk.y = 0.f; }
    float2 Xn = spec_bin(z, nk); if (!keep[nk]) { Xn.x = 0.f; Xn.y = 0.f; }
    float2 Xe = { 0.5f * (Xk.x + Xn.x), 0.5f * (Xk.y - Xn.y) };
    float2 Dd = { 0.5f * (Xk.x - Xn.x), 0.5f * (Xk.y + Xn.y) };
    float ss, cc;
    __sincosf(3.14159265358979f * k * (1.f / NC), &ss, &cc);
    float2 Xo = { Dd.x * cc - Dd.y * ss, Dd.x * ss + Dd.y * cc };
    z[k & (NC - 1)] = { Xe.x - Xo.y, -(Xe.y + Xo.x) };      // conj(Z'[k])
    if (k != 0 && k != NC / 2)
      z[nk] = { Xe.x + Xo.y, Xe.y - Xo.x };                 // conj(Z'[N-k])
  }
  __syncthreads();

  for (int i = t; i < NC; i += 256) {
    int r = __brev((unsigned)i) >> 20;
    if (r > i) { float2 tmp = z[i]; z[i] = z[r]; z[r] = tmp; }
  }
  __syncthreads();
  fft_stages(z, t);

  const float inv = 1.f / NC;
  for (int n = t; n < NC; n += 256) {
    float2 v = z[n];
    ((float2*)sig)[n] = make_float2(v.x * inv, -v.y * inv);
  }
}

// ---------------------------------------------------------------------------
// K3: transpose back (B,D,L)->(B,L,D), * filt_w, LayerNorm2, emit bf16 A
// ---------------------------------------------------------------------------
__global__ __launch_bounds__(256) void T_filt_ln2_kernel(
    const float* __restrict__ ft, const float* __restrict__ filt_w,
    const float* __restrict__ g2, const float* __restrict__ b2,
    u16* __restrict__ Abf) {
  __shared__ float tile[32][D_DIM + 1];
  const int b = blockIdx.y, l0 = blockIdx.x * 32, t = threadIdx.x;
  const float* src = ft + ((size_t)b * D_DIM + t) * L_DIM + l0;   // thread t == d
#pragma unroll
  for (int j = 0; j < 32; j += 4) {
    float4 v = *(const float4*)(src + j);
    tile[j][t] = v.x; tile[j + 1][t] = v.y; tile[j + 2][t] = v.z; tile[j + 3][t] = v.w;
  }
#pragma unroll 8
  for (int r = 0; r < 32; ++r)           // same-thread column, no barrier needed
    tile[r][t] *= filt_w[(size_t)(l0 + r) * D_DIM + t];
  __syncthreads();
  const int wid = t >> 5, lane = t & 31;
  for (int q = 0; q < 4; ++q) {
    const int r = wid * 4 + q;
    float s = 0.f, s2 = 0.f;
#pragma unroll
    for (int i = 0; i < 8; ++i) {
      float v = tile[r][lane + 32 * i];
      s += v; s2 += v * v;
    }
    s = wave_sum(s); s2 = wave_sum(s2);
    float mean = s * (1.f / D_DIM);
    float var  = s2 * (1.f / D_DIM) - mean * mean;
    float rstd = rsqrtf(var + 1e-5f);
    u16* dst = Abf + ((size_t)b * L_DIM + l0 + r) * D_DIM;
#pragma unroll
    for (int i = 0; i < 8; ++i) {
      int d = lane + 32 * i;
      float v = (tile[r][d] - mean) * rstd * g2[d] + b2[d];
      dst[d] = f2bf(v);
    }
  }
}

// ---------------------------------------------------------------------------
// WMMA GEMM. Fragment layout (bf16 16x16x32): lanes 0-15 own row/col
// (lane&15), K = {0..7,16..23}; lanes 16-31 own K = {8..15,24..31}.
// ---------------------------------------------------------------------------
#define PITCH 40   /* 32 K elems + 8 pad (16B-aligned rows, conflict-free) */

__device__ __forceinline__ v16bf load_frag(const u16* rowbase, int koff) {
  U4x2 p;
  p.a = *(const uint4*)(rowbase + koff);
  p.b = *(const uint4*)(rowbase + koff + 16);
  return __builtin_bit_cast(v16bf, p);
}

// stage one 128x32 bf16 tile (row-major, K contiguous) into LDS
__device__ __forceinline__ void stage_tile(u16* lds, const u16* gsrc_row,
                                           int k0, int lrow, int lpart) {
#ifdef HAVE_ASYNC_LDS
  ASYNC_CP16(gsrc_row + k0 + lpart,     &lds[lrow * PITCH + lpart]);
  ASYNC_CP16(gsrc_row + k0 + lpart + 8, &lds[lrow * PITCH + lpart + 8]);
#else
  *(uint4*)(&lds[lrow * PITCH + lpart])     = *(const uint4*)(gsrc_row + k0 + lpart);
  *(uint4*)(&lds[lrow * PITCH + lpart + 8]) = *(const uint4*)(gsrc_row + k0 + lpart + 8);
#endif
}

// GEMM1: (65536x256) x (256x1024) + bb1, exact GELU -> bf16. grid (8, 512)
__global__ __launch_bounds__(256) void gemm1_kernel(
    const u16* __restrict__ A, const u16* __restrict__ Bt,
    const float* __restrict__ bias, u16* __restrict__ Out) {
  __shared__ u16 As[128 * PITCH];
  __shared__ u16 Bs[128 * PITCH];
  const int t = threadIdx.x;
  const int m0 = blockIdx.y * 128, n0 = blockIdx.x * 128;
  const int wid = t >> 5, lane = t & 31;
  const int wm = (wid >> 2) * 64, wn = (wid & 3) * 32;
  const int lrow = t >> 1, lpart = (t & 1) * 16;
  const u16* Asrc = A  + (size_t)(m0 + lrow) * D_DIM;
  const u16* Bsrc = Bt + (size_t)(n0 + lrow) * D_DIM;
  v8f acc[4][2] = {};
  const int koff = (lane >> 4) * 8;
  for (int k0 = 0; k0 < D_DIM; k0 += 32) {
    __builtin_prefetch(Asrc + k0 + 32, 0, 3);
    __builtin_prefetch(Bsrc + k0 + 32, 0, 3);
    stage_tile(As, Asrc, k0, lrow, lpart);
    stage_tile(Bs, Bsrc, k0, lrow, lpart);
#ifdef HAVE_ASYNC_LDS
    async_wait0();
#endif
    __syncthreads();
    v16bf bfrag[2];
#pragma unroll
    for (int nt = 0; nt < 2; ++nt)
      bfrag[nt] = load_frag(&Bs[(wn + nt * 16 + (lane & 15)) * PITCH], koff);
#pragma unroll
    for (int mt = 0; mt < 4; ++mt) {
      v16bf afrag = load_frag(&As[(wm + mt * 16 + (lane & 15)) * PITCH], koff);
#pragma unroll
      for (int nt = 0; nt < 2; ++nt)
        acc[mt][nt] = __builtin_amdgcn_wmma_f32_16x16x32_bf16(
            false, afrag, false, bfrag[nt], (short)0, acc[mt][nt], false, false);
    }
    __syncthreads();
  }
  const int mbase = m0 + wm + 8 * (lane >> 4);
  const int nbase = n0 + wn + (lane & 15);
#pragma unroll
  for (int mt = 0; mt < 4; ++mt)
#pragma unroll
    for (int nt = 0; nt < 2; ++nt) {
      int n = nbase + nt * 16;
      float bn = bias[n];
      AccU cu; cu.v = acc[mt][nt];
#pragma unroll
      for (int r = 0; r < 8; ++r) {
        int m = mbase + mt * 16 + r;
        float v = cu.f[r] + bn;
        v = 0.5f * v * (1.f + erff(v * 0.70710678118654752f));
        Out[(size_t)m * H_DIM + n] = f2bf(v);
      }
    }
}

// GEMM2: (65536x1024) x (1024x256) + bb2 + residual -> f32 out. grid (2, 512)
__global__ __launch_bounds__(256) void gemm2_kernel(
    const u16* __restrict__ A, const u16* __restrict__ Bt,
    const float* __restrict__ bias, const float* __restrict__ resid,
    float* __restrict__ Out) {
  __shared__ u16 As[128 * PITCH];
  __shared__ u16 Bs[128 * PITCH];
  const int t = threadIdx.x;
  const int m0 = blockIdx.y * 128, n0 = blockIdx.x * 128;
  const int wid = t >> 5, lane = t & 31;
  const int wm = (wid >> 2) * 64, wn = (wid & 3) * 32;
  const int lrow = t >> 1, lpart = (t & 1) * 16;
  const u16* Asrc = A  + (size_t)(m0 + lrow) * H_DIM;
  const u16* Bsrc = Bt + (size_t)(n0 + lrow) * H_DIM;
  v8f acc[4][2] = {};
  const int koff = (lane >> 4) * 8;
  for (int k0 = 0; k0 < H_DIM; k0 += 32) {
    __builtin_prefetch(Asrc + k0 + 32, 0, 3);
    __builtin_prefetch(Bsrc + k0 + 32, 0, 3);
    stage_tile(As, Asrc, k0, lrow, lpart);
    stage_tile(Bs, Bsrc, k0, lrow, lpart);
#ifdef HAVE_ASYNC_LDS
    async_wait0();
#endif
    __syncthreads();
    v16bf bfrag[2];
#pragma unroll
    for (int nt = 0; nt < 2; ++nt)
      bfrag[nt] = load_frag(&Bs[(wn + nt * 16 + (lane & 15)) * PITCH], koff);
#pragma unroll
    for (int mt = 0; mt < 4; ++mt) {
      v16bf afrag = load_frag(&As[(wm + mt * 16 + (lane & 15)) * PITCH], koff);
#pragma unroll
      for (int nt = 0; nt < 2; ++nt)
        acc[mt][nt] = __builtin_amdgcn_wmma_f32_16x16x32_bf16(
            false, afrag, false, bfrag[nt], (short)0, acc[mt][nt], false, false);
    }
    __syncthreads();
  }
  const int mbase = m0 + wm + 8 * (lane >> 4);
  const int nbase = n0 + wn + (lane & 15);
#pragma unroll
  for (int mt = 0; mt < 4; ++mt)
#pragma unroll
    for (int nt = 0; nt < 2; ++nt) {
      int n = nbase + nt * 16;
      float bn = bias[n];
      AccU cu; cu.v = acc[mt][nt];
#pragma unroll
      for (int r = 0; r < 8; ++r) {
        int m = mbase + mt * 16 + r;
        size_t idx = (size_t)m * D_DIM + n;
        Out[idx] = cu.f[r] + bn + resid[idx];
      }
    }
}

// ---------------------------------------------------------------------------
extern "C" void kernel_launch(void* const* d_in, const int* in_sizes, int n_in,
                              void* d_out, int out_size, void* d_ws, size_t ws_size,
                              hipStream_t stream) {
  const float* x      = (const float*)d_in[0];
  const float* g1     = (const float*)d_in[1];
  const float* b1     = (const float*)d_in[2];
  const float* g2     = (const float*)d_in[3];
  const float* b2     = (const float*)d_in[4];
  const float* filt_w = (const float*)d_in[5];
  const float* w1     = (const float*)d_in[6];
  const float* bb1    = (const float*)d_in[7];
  const float* w2     = (const float*)d_in[8];
  const float* bb2    = (const float*)d_in[9];
  float* out = (float*)d_out;

  char* ws = (char*)d_ws;
  float* ht  = (float*)ws;                                   // 64 MiB (B,D,L), FFT in place
  u16*   Abf = (u16*)(ws + ((size_t)64 << 20));              // 32 MiB bf16 (B*L, D)
  u16*   Mm  = (u16*)(ws + ((size_t)96 << 20));              // 128 MiB bf16 (B*L, H)
  u16*   W1t = (u16*)(ws + ((size_t)224 << 20));             // 0.5 MiB
  u16*   W2t = (u16*)(ws + ((size_t)224 << 20) + (1 << 19)); // 0.5 MiB

  prep_w_kernel<<<dim3((D_DIM * H_DIM + 255) / 256), 256, 0, stream>>>(w1, w2, W1t, W2t);
  ln1_relu_T_kernel<<<dim3(L_DIM / 32, B_DIM), 256, 0, stream>>>(x, g1, b1, ht);
  fft_filter_kernel<<<dim3(B_DIM * D_DIM), 256, 0, stream>>>(ht);
  T_filt_ln2_kernel<<<dim3(L_DIM / 32, B_DIM), 256, 0, stream>>>(ht, filt_w, g2, b2, Abf);
  gemm1_kernel<<<dim3(H_DIM / 128, (B_DIM * L_DIM) / 128), 256, 0, stream>>>(Abf, W1t, bb1, Mm);
  gemm2_kernel<<<dim3(D_DIM / 128, (B_DIM * L_DIM) / 128), 256, 0, stream>>>(Mm, W2t, bb2, x, out);
}
